// MixtureOfExperts_72438918414758
// MI455X (gfx1250) — compile-verified, matrix operand order checked by reading, not run on hardware
//
#include <hip/hip_runtime.h>
#include <hip/hip_bf16.h>
#include <math.h>

// MoE (top-2 of 8 experts), H=768, I=3072, T=2048 tokens.
// bf16 WMMA (v_wmma_f32_16x16x32_bf16) with fp32 accumulate; gather-based
// expert execution (exactly K*T = 4096 assignment rows of GEMM work).
// GEMM: 64x128 block tile, 8 waves (4M x 2N), 16x64 per wave (4 WMMA accums),
// double-buffered LDS weight tile (fp32 -> bf16 on the fly), 1 barrier/K-step.

typedef __bf16 bf16_t;
typedef __attribute__((ext_vector_type(16))) __bf16 v16bf;
typedef __attribute__((ext_vector_type(8)))  __bf16 v8bf;
typedef __attribute__((ext_vector_type(8)))  float  v8f;

#define T_TOK 2048
#define H_DIM 768
#define I_DIM 3072
#define E_NUM 8
#define A_TOT (T_TOK * 2)
#define LN_EPS 1e-5f

#define TILE_N 128
#define LDS_STRIDE 40   // 32 k-elems + 8 pad (keeps 16B alignment, spreads banks)

__device__ __forceinline__ bf16_t f2bf(float f) {
  unsigned u = __builtin_bit_cast(unsigned, f);
  u += 0x7FFFu + ((u >> 16) & 1u);                 // round-to-nearest-even
  unsigned short hs = (unsigned short)(u >> 16);
  return __builtin_bit_cast(bf16_t, hs);
}

// ---------------- x -> bf16 ----------------
__global__ void cvt_x_kernel(const float* __restrict__ x, bf16_t* __restrict__ xb) {
  int n = T_TOK * H_DIM;
  for (int i = blockIdx.x * blockDim.x + threadIdx.x; i < n; i += gridDim.x * blockDim.x)
    xb[i] = f2bf(x[i]);
}

// ---------------- router: one wave per token ----------------
__global__ __launch_bounds__(256) void moe_router(
    const float* __restrict__ x, const float* __restrict__ Wg,
    const float* __restrict__ bg, int* __restrict__ top_e, float* __restrict__ top_w) {
  int wave = threadIdx.x >> 5;
  int lane = threadIdx.x & 31;
  int t = blockIdx.x * 8 + wave;
  if (t >= T_TOK) return;
  float acc[E_NUM];
#pragma unroll
  for (int e = 0; e < E_NUM; ++e) acc[e] = 0.f;
  const float* xr = x + (size_t)t * H_DIM;
  for (int h = lane; h < H_DIM; h += 32) {
    float xv = xr[h];
    const float* wr = Wg + (size_t)h * E_NUM;
#pragma unroll
    for (int e = 0; e < E_NUM; ++e) acc[e] += xv * wr[e];
  }
#pragma unroll
  for (int e = 0; e < E_NUM; ++e) {
#pragma unroll
    for (int off = 16; off > 0; off >>= 1) acc[e] += __shfl_xor(acc[e], off, 32);
  }
  if (lane == 0) {
    float p[E_NUM];
    float mx = -1e30f;
#pragma unroll
    for (int e = 0; e < E_NUM; ++e) { p[e] = acc[e] + bg[e]; mx = fmaxf(mx, p[e]); }
    float sum = 0.f;
#pragma unroll
    for (int e = 0; e < E_NUM; ++e) { p[e] = expf(p[e] - mx); sum += p[e]; }
    float inv = 1.0f / sum;
#pragma unroll
    for (int e = 0; e < E_NUM; ++e) p[e] *= inv;
    int i0 = 0; float v0 = p[0];
#pragma unroll
    for (int e = 1; e < E_NUM; ++e) if (p[e] > v0) { v0 = p[e]; i0 = e; }   // first index wins ties
    int i1 = -1; float v1 = -1.0f;
#pragma unroll
    for (int e = 0; e < E_NUM; ++e) if (e != i0 && p[e] > v1) { v1 = p[e]; i1 = e; }
    float s2 = v0 + v1;
    top_e[2 * t] = i0;       top_e[2 * t + 1] = i1;
    top_w[2 * t] = v0 / s2;  top_w[2 * t + 1] = v1 / s2;
  }
}

// ---------------- deterministic partition: one wave per expert ----------------
__global__ __launch_bounds__(256) void moe_partition(
    const int* __restrict__ top_e, const float* __restrict__ top_w,
    int* __restrict__ counts, int* __restrict__ offs,
    int* __restrict__ rowtok, float* __restrict__ gate, int* __restrict__ slot_of) {
  int e = threadIdx.x >> 5;
  int lane = threadIdx.x & 31;
  int cnt = 0;
  for (int base = 0; base < T_TOK; base += 32) {
    int t = base + lane;
    bool sel = (top_e[2 * t] == e) || (top_e[2 * t + 1] == e);
    unsigned long long b = __ballot(sel);
    cnt += __popc((unsigned)(b & 0xffffffffull));
  }
  if (lane == 0) counts[e] = cnt;
  __syncthreads();
  if (threadIdx.x == 0) {
    int o = 0;
    for (int i = 0; i < E_NUM; ++i) { offs[i] = o; o += counts[i]; }
  }
  __syncthreads();
  int pos = offs[e];
  unsigned lt = (lane == 0) ? 0u : ((1u << lane) - 1u);
  for (int base = 0; base < T_TOK; base += 32) {
    int t = base + lane;
    int e0 = top_e[2 * t], e1 = top_e[2 * t + 1];
    bool sel = (e0 == e) || (e1 == e);
    unsigned long long b = __ballot(sel);
    unsigned m = (unsigned)(b & 0xffffffffull);
    if (sel) {
      int slot = pos + __popc(m & lt);
      int k = (e0 == e) ? 0 : 1;
      rowtok[slot] = t;
      gate[slot] = top_w[2 * t + k];
      slot_of[2 * t + k] = slot;
    }
    pos += __popc(m);
  }
}

// ---------------- WMMA GEMM: C[rows_e x Ndim] = A[rows] x W_e[Kdim x Ndim] ----------------
template <bool GELU_OUT>
__global__ __launch_bounds__(256) void moe_gemm(
    const bf16_t* __restrict__ A, int lda,
    const float* __restrict__ Wfull, const float* __restrict__ bias,
    int Kdim, int Ndim,
    const int* __restrict__ counts, const int* __restrict__ offs,
    const int* __restrict__ rowtok,              // null -> identity (slot rows)
    bf16_t* __restrict__ Hout, float* __restrict__ Yout) {
  const int e = blockIdx.z;
  const int rows = counts[e];
  const int m_blk = blockIdx.y * 64;
  if (m_blk >= rows) return;                     // block-uniform exit
  const int n_blk = blockIdx.x * TILE_N;
  const int wave = threadIdx.x >> 5;
  const int lane = threadIdx.x & 31;
  const int wm = wave >> 1;                      // 0..3 (M, 16 each)
  const int wn = wave & 1;                       // 0..1 (N, 64 each)
  const int lane15 = lane & 15;
  const int khalf = (lane >> 4) * 8;             // 0 or 8 per WMMA 16-bit layout

  // double-buffered weight tile: [n][k] layout, bf16
  __shared__ __align__(16) bf16_t Bs[2][TILE_N * LDS_STRIDE];

  const float* W = Wfull + (size_t)e * Kdim * Ndim;
  const int base = offs[e];

  int mloc = m_blk + wm * 16 + lane15;
  if (mloc >= rows) mloc = rows - 1;             // clamp (stores are guarded)
  const int arow = rowtok ? rowtok[base + mloc] : (base + mloc);
  const bf16_t* Arow = A + (size_t)arow * lda;

  // cooperative weight-tile loader: thread covers (n = tid&127, k chunk of 16)
  const int cn = threadIdx.x & 127;
  const int ck = (threadIdx.x >> 7) * 16;        // 0 or 16

  v8f c[4] = {v8f{}, v8f{}, v8f{}, v8f{}};

  // prologue: tile k0 = 0 -> Bs[0]
  {
    const float* wp = W + (size_t)ck * Ndim + (n_blk + cn);
    v8bf p0, p1;
#pragma unroll
    for (int j = 0; j < 8; ++j) p0[j] = f2bf(wp[(size_t)j * Ndim]);
#pragma unroll
    for (int j = 0; j < 8; ++j) p1[j] = f2bf(wp[(size_t)(j + 8) * Ndim]);
    *(v8bf*)&Bs[0][cn * LDS_STRIDE + ck] = p0;
    *(v8bf*)&Bs[0][cn * LDS_STRIDE + ck + 8] = p1;
  }
  __syncthreads();

  int buf = 0;
  for (int k0 = 0; k0 < Kdim; k0 += 32) {
    const bool has_next = (k0 + 32 < Kdim);
    // issue next tile's global loads early (overlap with WMMA)
    float nxt[16];
    if (has_next) {
      const float* wp = W + (size_t)(k0 + 32 + ck) * Ndim + (n_blk + cn);
#pragma unroll
      for (int j = 0; j < 16; ++j) nxt[j] = wp[(size_t)j * Ndim];
    }

    // A fragment: row = lane15, K = khalf+[0..7] and khalf+16+[0..7]
    v16bf af;
    {
      const bf16_t* ap = Arow + k0 + khalf;
      v8bf lo = *(const v8bf*)(ap);
      v8bf hi = *(const v8bf*)(ap + 16);
#pragma unroll
      for (int j = 0; j < 8; ++j) { af[j] = lo[j]; af[8 + j] = hi[j]; }
    }
    // 4 B fragments + 4 WMMAs (wave's 16x64 output)
#pragma unroll
    for (int sub = 0; sub < 4; ++sub) {
      const bf16_t* bp = &Bs[buf][(wn * 64 + sub * 16 + lane15) * LDS_STRIDE + khalf];
      v8bf lo = *(const v8bf*)(bp);
      v8bf hi = *(const v8bf*)(bp + 16);
      v16bf bf;
#pragma unroll
      for (int j = 0; j < 8; ++j) { bf[j] = lo[j]; bf[8 + j] = hi[j]; }
      c[sub] = __builtin_amdgcn_wmma_f32_16x16x32_bf16(false, af, false, bf, (short)0,
                                                       c[sub], false, false);
    }

    if (has_next) {  // convert + store next tile into the other buffer
      v8bf p0, p1;
#pragma unroll
      for (int j = 0; j < 8; ++j) p0[j] = f2bf(nxt[j]);
#pragma unroll
      for (int j = 0; j < 8; ++j) p1[j] = f2bf(nxt[j + 8]);
      *(v8bf*)&Bs[buf ^ 1][cn * LDS_STRIDE + ck] = p0;
      *(v8bf*)&Bs[buf ^ 1][cn * LDS_STRIDE + ck + 8] = p1;
    }
    __syncthreads();
    buf ^= 1;
  }

  // store: D rows = j + (lane>=16 ? 8 : 0), col = lane&15
  const float* bp = bias + (size_t)e * Ndim;
  const int mrow0 = m_blk + wm * 16 + ((lane < 16) ? 0 : 8);
#pragma unroll
  for (int sub = 0; sub < 4; ++sub) {
    int n = n_blk + wn * 64 + sub * 16 + lane15;
    float bv = bp[n];
#pragma unroll
    for (int j = 0; j < 8; ++j) {
      int m = mrow0 + j;
      if (m < rows) {
        float v = c[sub][j] + bv;
        size_t orow = (size_t)(base + m);
        if constexpr (GELU_OUT) {
          v = 0.5f * v * (1.0f + erff(v * 0.70710678118654752f));  // exact GELU
          Hout[orow * (size_t)Ndim + n] = f2bf(v);
        } else {
          Yout[orow * (size_t)Ndim + n] = v;
        }
      }
    }
  }
}

// ---------------- combine (fixed order -> deterministic) + residual + LayerNorm ----------------
__global__ __launch_bounds__(256) void moe_combine_ln(
    const float* __restrict__ x, const float* __restrict__ y,
    const float* __restrict__ gate, const int* __restrict__ slot_of,
    const float* __restrict__ gamma, const float* __restrict__ beta,
    float* __restrict__ out) {
  int t = blockIdx.x;
  int s0 = slot_of[2 * t], s1 = slot_of[2 * t + 1];
  float w0 = gate[s0], w1 = gate[s1];
  const float* xr = x + (size_t)t * H_DIM;
  const float* y0 = y + (size_t)s0 * H_DIM;
  const float* y1 = y + (size_t)s1 * H_DIM;
  float z[3];
  float s = 0.f, sq = 0.f;
#pragma unroll
  for (int i = 0; i < 3; ++i) {
    int hh = threadIdx.x + i * 256;
    float v = xr[hh] + w0 * y0[hh] + w1 * y1[hh];
    z[i] = v; s += v; sq += v * v;
  }
#pragma unroll
  for (int off = 16; off > 0; off >>= 1) {
    s += __shfl_xor(s, off, 32);
    sq += __shfl_xor(sq, off, 32);
  }
  __shared__ float rs[8], rq[8];
  int wave = threadIdx.x >> 5, lane = threadIdx.x & 31;
  if (lane == 0) { rs[wave] = s; rq[wave] = sq; }
  __syncthreads();
  if (threadIdx.x == 0) {
    float S = 0.f, Q = 0.f;
    for (int i = 0; i < 8; ++i) { S += rs[i]; Q += rq[i]; }
    rs[0] = S; rq[0] = Q;
  }
  __syncthreads();
  float mean = rs[0] * (1.0f / H_DIM);
  float var = rq[0] * (1.0f / H_DIM) - mean * mean;
  float inv = rsqrtf(var + LN_EPS);
#pragma unroll
  for (int i = 0; i < 3; ++i) {
    int hh = threadIdx.x + i * 256;
    out[(size_t)t * H_DIM + hh] = (z[i] - mean) * inv * gamma[hh] + beta[hh];
  }
}

extern "C" void kernel_launch(void* const* d_in, const int* in_sizes, int n_in,
                              void* d_out, int out_size, void* d_ws, size_t ws_size,
                              hipStream_t stream) {
  const float* x  = (const float*)d_in[0];
  const float* W1 = (const float*)d_in[1];
  const float* b1 = (const float*)d_in[2];
  const float* W2 = (const float*)d_in[3];
  const float* b2 = (const float*)d_in[4];
  const float* Wg = (const float*)d_in[5];
  const float* bg = (const float*)d_in[6];
  const float* gamma = (const float*)d_in[7];
  const float* beta  = (const float*)d_in[8];
  float* out = (float*)d_out;

  char* ws = (char*)d_ws;
  size_t off = 0;
  auto alloc = [&](size_t bytes) -> void* {
    void* p = ws + off;
    off = (off + bytes + 255) & ~(size_t)255;
    return p;
  };
  bf16_t* xb     = (bf16_t*)alloc((size_t)T_TOK * H_DIM * sizeof(bf16_t));  // 3.1 MB
  bf16_t* hbuf   = (bf16_t*)alloc((size_t)A_TOT * I_DIM * sizeof(bf16_t));  // 25.2 MB
  float*  ybuf   = (float*) alloc((size_t)A_TOT * H_DIM * sizeof(float));   // 12.6 MB
  int*    top_e  = (int*)   alloc((size_t)T_TOK * 2 * sizeof(int));
  float*  top_w  = (float*) alloc((size_t)T_TOK * 2 * sizeof(float));
  int*    counts = (int*)   alloc(E_NUM * sizeof(int));
  int*    offsv  = (int*)   alloc(E_NUM * sizeof(int));
  int*    rowtok = (int*)   alloc((size_t)A_TOT * sizeof(int));
  float*  gate   = (float*) alloc((size_t)A_TOT * sizeof(float));
  int*    slotof = (int*)   alloc((size_t)T_TOK * 2 * sizeof(int));

  cvt_x_kernel<<<1536, 256, 0, stream>>>(x, xb);
  moe_router<<<T_TOK / 8, 256, 0, stream>>>(x, Wg, bg, top_e, top_w);
  moe_partition<<<1, 256, 0, stream>>>(top_e, top_w, counts, offsv, rowtok, gate, slotof);
  // GEMM1 + GELU: A = xb[token], W1_e [H x I], out bf16 h [slot][I]
  moe_gemm<true><<<dim3(I_DIM / TILE_N, (T_TOK + 63) / 64, E_NUM), 256, 0, stream>>>(
      xb, H_DIM, W1, b1, H_DIM, I_DIM, counts, offsv, rowtok, hbuf, nullptr);
  // GEMM2: A = h[slot], W2_e [I x H], out fp32 y [slot][H]
  moe_gemm<false><<<dim3(H_DIM / TILE_N, (T_TOK + 63) / 64, E_NUM), 256, 0, stream>>>(
      hbuf, I_DIM, W2, b2, I_DIM, H_DIM, counts, offsv, nullptr, nullptr, ybuf);
  moe_combine_ln<<<T_TOK, 256, 0, stream>>>(x, ybuf, gate, slotof, gamma, beta, out);
}
